// MyGNN_9964324127252
// MI455X (gfx1250) — matrix operand order repaired and checked by reading, not compiled, not run on hardware
//
#include <hip/hip_runtime.h>

// ---------------------------------------------------------------------------
// HGT (heterogeneous graph transformer) forward on MI455X / gfx1250.
//
// Dense node-feature GEMMs run through v_wmma_f32_16x16x32_bf16 (wave32,
// 16x16 tiles, K-step 32). Activations are pre-converted to bf16 once per
// layer (each matrix feeds 3 GEMMs), so the GEMM inner loop is pure
// b128-load + wmma with no packing VALU. Per-head relation transforms
// (w_att / w_msg) are algebraically fused into the k/v projection weights
// on-device. Edge softmax / aggregation are atomic + gather bound and stay
// in scalar fp32 kernels.
// ---------------------------------------------------------------------------

typedef __bf16 v16bf __attribute__((ext_vector_type(16)));
typedef float  v8f   __attribute__((ext_vector_type(8)));

#define D_    256
#define H_    8
#define DH_   32
#define DIN_  128
#define DOUT_ 16
#define L_    2

static __device__ __forceinline__ unsigned bfbits(float f) {
  unsigned u = __float_as_uint(f);
  return (u + 0x7FFFu + ((u >> 16) & 1u)) >> 16;   // round-to-nearest-even
}
static __device__ __forceinline__ unsigned pack2(float lo, float hi) {
  return bfbits(lo) | (bfbits(hi) << 16);
}
// monotone float <-> uint mapping for atomicMax-based segment max
static __device__ __forceinline__ unsigned mapf(float f) {
  unsigned u = __float_as_uint(f);
  return (u & 0x80000000u) ? ~u : (u | 0x80000000u);
}
static __device__ __forceinline__ float unmapf(unsigned m) {
  unsigned u = (m & 0x80000000u) ? (m & 0x7FFFFFFFu) : ~m;
  return __uint_as_float(u);
}

// ---------------------------------------------------------------------------
// WMMA GEMM:  C[M,N] = A[M,K] (bf16, row-major, leading dim lda elements)
//                      @ Bp    (bf16, packed fragment-native layout)
//                      + bias[N]
// One wave -> 16 x (16*NT) output tile; the 4 waves of a block share one
// M-tile (same A rows) across the N groups. K multiple of 32, N multiple of
// 16*NT, M multiple of 16 (true for all call sites: M=50000, K in {128,256}).
// ---------------------------------------------------------------------------
template <int NT>
__global__ __launch_bounds__(128)
void wmma_gemm(const unsigned short* __restrict__ A, int lda,
               const unsigned* __restrict__ Bp,
               const float* __restrict__ bias,
               float* __restrict__ C, int ldc,
               int M, int N, int K) {
  const int lane = threadIdx.x & 31;
  const int wv   = threadIdx.x >> 5;
  const int nGroups = N / (16 * NT);
  const int w      = blockIdx.x * 4 + wv;
  const int mTile  = w / nGroups;
  const int nGroup = w % nGroups;
  if (mTile * 16 >= M) return;

  const int half = lane >> 4;     // which 16-lane half of the wave
  const int l16  = lane & 15;
  const size_t row = (size_t)(mTile * 16 + l16);
  const int nKs = K >> 5;

  v8f acc[NT];
#pragma unroll
  for (int j = 0; j < NT; ++j)
#pragma unroll
    for (int r = 0; r < 8; ++r) acc[j][r] = 0.0f;

  // A fragment (16x32 bf16): lanes 0-15 rows M, K octets {0-7,16-23};
  // lanes 16-31 same rows, K octets {8-15,24-31}. bf16 row-major matches the
  // fragment layout: two coalesced b128 loads per K-step, zero packing ALU.
  const unsigned short* arow = A + row * (size_t)lda + half * 8;

  for (int ks = 0; ks < nKs; ++ks) {
    const uint4* ap = (const uint4*)(arow + ks * 32);
    union { v16bf v; uint4 q[2]; } av;
    av.q[0] = ap[0];               // K = kb + half*8 + 0..7
    av.q[1] = ap[2];               // K = kb + 16 + half*8 + 0..7

    const unsigned* bbase =
        Bp + ((size_t)(nGroup * NT) * nKs + ks) * 256 + lane * 8;
#pragma unroll
    for (int j = 0; j < NT; ++j) {
      const uint4* bp = (const uint4*)(bbase + (size_t)j * nKs * 256);
      union { v16bf v; uint4 q[2]; } bvv;
      bvv.q[0] = bp[0]; bvv.q[1] = bp[1];
      acc[j] = __builtin_amdgcn_wmma_f32_16x16x32_bf16(
          false, av.v, false, bvv.v, (short)0, acc[j], false, false);
    }
  }

  // C/D layout: lanes 0-15 -> rows r, lanes 16-31 -> rows r+8; col = l16.
#pragma unroll
  for (int j = 0; j < NT; ++j) {
    int ncol = (nGroup * NT + j) * 16 + l16;
    float b = bias ? bias[ncol] : 0.0f;
    float* cp = C + (size_t)(mTile * 16 + half * 8) * ldc + ncol;
#pragma unroll
    for (int r = 0; r < 8; ++r) cp[(size_t)r * ldc] = acc[j][r] + b;
  }
}

// fp32 -> bf16 activation conversion, 8 elements per thread (uint4 store).
__global__ void f32_to_bf16(const float* __restrict__ src,
                            unsigned short* __restrict__ dst, size_t n) {
  size_t i = ((size_t)blockIdx.x * blockDim.x + threadIdx.x) * 8;
  if (i >= n) return;
  const float4* s = (const float4*)(src + i);
  float4 a = s[0], b = s[1];
  uint4 o;
  o.x = pack2(a.x, a.y);
  o.y = pack2(a.z, a.w);
  o.z = pack2(b.x, b.y);
  o.w = pack2(b.z, b.w);
  *(uint4*)(dst + i) = o;
}

// Convert fp32 K x N weight matrix into fragment-native packed-bf16 layout:
// dword index = ((nt*nKs + ks)*32 + lane)*8 + v ; lane: n = nt*16 + (lane&15),
// kb = ks*32 + (lane>>4)*8 + (v>>2)*16 ; dword v holds bf16 pair (K = kb+2(v&3), +1).
__global__ void pack_b(const float* __restrict__ B, unsigned* __restrict__ Bp,
                       int K, int N) {
  int idx = blockIdx.x * 256 + threadIdx.x;
  int total = (K * N) >> 1;
  if (idx >= total) return;
  int v    = idx & 7;
  int lane = (idx >> 3) & 31;
  int frag = idx >> 8;
  int nKs  = K >> 5;
  int ks   = frag % nKs;
  int nt   = frag / nKs;
  int n  = nt * 16 + (lane & 15);
  int kb = ks * 32 + (lane >> 4) * 8 + (v >> 2) * 16 + (v & 3) * 2;
  float lo = B[(size_t)kb * N + n];
  float hi = B[(size_t)(kb + 1) * N + n];
  Bp[idx] = pack2(lo, hi);
}

// F[i, h*32+e] = sum_d W[i, h*32+d] * R[h, d, e]   (fold w_att/w_msg into Wk/Wv)
__global__ void fuse_w(const float* __restrict__ W, const float* __restrict__ R,
                       float* __restrict__ F) {
  int idx = blockIdx.x * 256 + threadIdx.x;   // 65536 total
  int i = idx >> 8;
  int c = idx & 255;
  int h = c >> 5, e = c & 31;
  const float* wr = W + (size_t)i * 256 + h * 32;
  const float* rr = R + h * 1024 + e;
  float s = 0.0f;
#pragma unroll
  for (int d = 0; d < 32; ++d) s += wr[d] * rr[d * 32];
  F[idx] = s;
}

__global__ void fuse_b(const float* __restrict__ b, const float* __restrict__ R,
                       float* __restrict__ F) {
  int c = threadIdx.x;   // 256 total
  int h = c >> 5, e = c & 31;
  float s = 0.0f;
#pragma unroll
  for (int d = 0; d < 32; ++d) s += b[h * 32 + d] * R[h * 1024 + d * 32 + e];
  F[c] = s;
}

__global__ void fill_f32(float* __restrict__ p, float v, size_t n) {
  size_t i = (size_t)blockIdx.x * blockDim.x + threadIdx.x;
  if (i < n) p[i] = v;
}
__global__ void fill_u32(unsigned* __restrict__ p, unsigned v, size_t n) {
  size_t i = (size_t)blockIdx.x * blockDim.x + threadIdx.x;
  if (i < n) p[i] = v;
}

// sc[e,h] = <kp[src,h,:], q[dst,h,:]> * mu[h] / sqrt(DH); segment max via
// monotone-uint atomicMax.
__global__ void edge_score(const float* __restrict__ kp,
                           const float* __restrict__ q,
                           const int* __restrict__ src,
                           const int* __restrict__ dst,
                           const float* __restrict__ mu,
                           float* __restrict__ sc,
                           unsigned* __restrict__ mmax, int E) {
  size_t idx = (size_t)blockIdx.x * blockDim.x + threadIdx.x;
  if (idx >= (size_t)E * H_) return;
  int e = (int)(idx >> 3);
  int h = (int)(idx & 7);
  int s = src[e], d = dst[e];
  const float4* kr = (const float4*)(kp + (size_t)s * D_ + h * DH_);
  const float4* qr = (const float4*)(q + (size_t)d * D_ + h * DH_);
  float acc = 0.0f;
#pragma unroll
  for (int j = 0; j < 8; ++j) {
    float4 a = kr[j], b = qr[j];
    acc += a.x * b.x + a.y * b.y + a.z * b.z + a.w * b.w;
  }
  float scv = acc * mu[h] * 0.17677669529663687f;   // 1/sqrt(32)
  sc[idx] = scv;
  atomicMax(&mmax[(size_t)d * H_ + h], mapf(scv));
}

// ex = exp(sc - max[dst]); segment sum of ex into den.
__global__ void edge_expsum(float* __restrict__ sc,
                            const unsigned* __restrict__ mmax,
                            const int* __restrict__ dst,
                            float* __restrict__ den, int E) {
  size_t idx = (size_t)blockIdx.x * blockDim.x + threadIdx.x;
  if (idx >= (size_t)E * H_) return;
  int e = (int)(idx >> 3);
  int h = (int)(idx & 7);
  int d = dst[e];
  float m = unmapf(mmax[(size_t)d * H_ + h]);
  float x = __expf(sc[idx] - m);
  sc[idx] = x;
  atomicAdd(&den[(size_t)d * H_ + h], x);
}

// agg[dst,h,:] += (ex/den[dst,h]) * vp[src,h,:]; 64 threads per edge,
// 4 floats per thread.
__global__ void edge_agg(const float* __restrict__ sc,
                         const float* __restrict__ den,
                         const float* __restrict__ vp,
                         const int* __restrict__ src,
                         const int* __restrict__ dst,
                         float* __restrict__ agg, int E) {
  size_t idx = (size_t)blockIdx.x * blockDim.x + threadIdx.x;
  if (idx >= (size_t)E * 64) return;
  int e  = (int)(idx >> 6);
  int t  = (int)(idx & 63);
  int h  = t >> 3;
  int d4 = (t & 7) * 4;
  int s = src[e], d = dst[e];
  float a = sc[(size_t)e * H_ + h] / den[(size_t)d * H_ + h];
  float4 v = *(const float4*)(vp + (size_t)s * D_ + h * DH_ + d4);
  float* out = agg + (size_t)d * D_ + h * DH_ + d4;
  atomicAdd(out + 0, a * v.x);
  atomicAdd(out + 1, a * v.y);
  atomicAdd(out + 2, a * v.z);
  atomicAdd(out + 3, a * v.w);
}

// ---------------------------------------------------------------------------
static inline void gemm_launch(const unsigned short* A, int lda,
                               const unsigned* Bp, const float* bias,
                               float* C, int ldc,
                               int M, int N, int K, hipStream_t s) {
  int mT = (M + 15) / 16;
  if (N % 64 == 0) {
    int waves = mT * (N / 64);
    int blocks = (waves + 3) / 4;
    wmma_gemm<4><<<blocks, 128, 0, s>>>(A, lda, Bp, bias, C, ldc, M, N, K);
  } else {
    int waves = mT * (N / 16);
    int blocks = (waves + 3) / 4;
    wmma_gemm<1><<<blocks, 128, 0, s>>>(A, lda, Bp, bias, C, ldc, M, N, K);
  }
}

static inline void conv_launch(const float* src, unsigned short* dst, size_t n,
                               hipStream_t s) {
  f32_to_bf16<<<(n / 8 + 255) / 256, 256, 0, s>>>(src, dst, n);
}

extern "C" void kernel_launch(void* const* d_in, const int* in_sizes, int n_in,
                              void* d_out, int out_size, void* d_ws,
                              size_t ws_size, hipStream_t stream) {
  (void)n_in; (void)out_size; (void)ws_size;

  const float* feat_user = (const float*)d_in[0];
  const float* feat_item = (const float*)d_in[1];
  const float* Win_user  = (const float*)d_in[2];
  const float* bin_user  = (const float*)d_in[3];
  const float* Win_item  = (const float*)d_in[4];
  const float* bin_item  = (const float*)d_in[5];
  const float* Wk        = (const float*)d_in[6];
  const float* bk        = (const float*)d_in[7];
  const float* Wq        = (const float*)d_in[8];
  const float* bq        = (const float*)d_in[9];
  const float* Wv        = (const float*)d_in[10];
  const float* bv        = (const float*)d_in[11];
  const float* w_att     = (const float*)d_in[12];
  const float* w_msg     = (const float*)d_in[13];
  const float* mu        = (const float*)d_in[14];
  const float* Wc        = (const float*)d_in[15];
  const float* bc        = (const float*)d_in[16];
  const int*   src_ui    = (const int*)d_in[17];
  const int*   dst_ui    = (const int*)d_in[18];
  const int*   src_iu    = (const int*)d_in[19];
  const int*   dst_iu    = (const int*)d_in[20];

  const int NU = in_sizes[0] / DIN_;
  const int NI = in_sizes[1] / DIN_;
  const int E  = in_sizes[17];
  const int NMAX = NU > NI ? NU : NI;

  // ---- carve workspace ----------------------------------------------------
  char* base = (char*)d_ws;
  size_t off = 0;
  auto takef = [&](size_t nelem) -> float* {
    float* p = (float*)(base + off);
    off = (off + nelem * 4 + 255) & ~(size_t)255;
    return p;
  };
  auto takeu = [&](size_t nelem) -> unsigned* {
    unsigned* p = (unsigned*)(base + off);
    off = (off + nelem * 4 + 255) & ~(size_t)255;
    return p;
  };
  auto takeh = [&](size_t nelem) -> unsigned short* {   // bf16 storage
    unsigned short* p = (unsigned short*)(base + off);
    off = (off + nelem * 2 + 255) & ~(size_t)255;
    return p;
  };

  float* hu  = takef((size_t)NU * D_);
  float* hi  = takef((size_t)NI * D_);
  float* kpu = takef((size_t)NU * D_);
  float* vpu = takef((size_t)NU * D_);
  float* qu  = takef((size_t)NU * D_);
  float* kpi = takef((size_t)NI * D_);
  float* vpi = takef((size_t)NI * D_);
  float* qi  = takef((size_t)NI * D_);
  float* sc  = takef((size_t)E * H_);
  unsigned* mmax = takeu((size_t)NMAX * H_);
  float* den = takef((size_t)NMAX * H_);
  float* FK  = takef((size_t)4 * D_ * D_);     // fused Wk·w_att  per (l,x)
  float* FV  = takef((size_t)4 * D_ * D_);     // fused Wv·w_msg  per (l,x)
  float* fbk = takef((size_t)4 * D_);
  float* fbv = takef((size_t)4 * D_);
  unsigned short* fub = takeh((size_t)NU * DIN_);   // bf16 feat_user
  unsigned short* fib = takeh((size_t)NI * DIN_);   // bf16 feat_item
  unsigned short* hub = takeh((size_t)NU * D_);     // bf16 hu
  unsigned short* hib = takeh((size_t)NI * D_);     // bf16 hi
  unsigned* WinUp = takeu((size_t)DIN_ * D_ / 2);
  unsigned* WinIp = takeu((size_t)DIN_ * D_ / 2);
  unsigned* Wqp   = takeu((size_t)4 * D_ * D_ / 2);
  unsigned* FKp   = takeu((size_t)4 * D_ * D_ / 2);
  unsigned* FVp   = takeu((size_t)4 * D_ * D_ / 2);
  unsigned* Wcp   = takeu((size_t)D_ * DOUT_ / 2);

  // ---- fuse relation transforms into projection weights -------------------
  for (int ix = 0; ix < 4; ++ix) {   // ix = l*2 + x
    fuse_w<<<256, 256, 0, stream>>>(Wk + (size_t)ix * D_ * D_,
                                    w_att + (size_t)ix * H_ * DH_ * DH_,
                                    FK + (size_t)ix * D_ * D_);
    fuse_w<<<256, 256, 0, stream>>>(Wv + (size_t)ix * D_ * D_,
                                    w_msg + (size_t)ix * H_ * DH_ * DH_,
                                    FV + (size_t)ix * D_ * D_);
    fuse_b<<<1, 256, 0, stream>>>(bk + (size_t)ix * D_,
                                  w_att + (size_t)ix * H_ * DH_ * DH_,
                                  fbk + (size_t)ix * D_);
    fuse_b<<<1, 256, 0, stream>>>(bv + (size_t)ix * D_,
                                  w_msg + (size_t)ix * H_ * DH_ * DH_,
                                  fbv + (size_t)ix * D_);
  }

  // ---- pack all weights into fragment-native bf16 -------------------------
  pack_b<<<(DIN_ * D_ / 2 + 255) / 256, 256, 0, stream>>>(Win_user, WinUp, DIN_, D_);
  pack_b<<<(DIN_ * D_ / 2 + 255) / 256, 256, 0, stream>>>(Win_item, WinIp, DIN_, D_);
  for (int ix = 0; ix < 4; ++ix) {
    pack_b<<<(D_ * D_ / 2 + 255) / 256, 256, 0, stream>>>(
        Wq + (size_t)ix * D_ * D_, Wqp + (size_t)ix * D_ * D_ / 2, D_, D_);
    pack_b<<<(D_ * D_ / 2 + 255) / 256, 256, 0, stream>>>(
        FK + (size_t)ix * D_ * D_, FKp + (size_t)ix * D_ * D_ / 2, D_, D_);
    pack_b<<<(D_ * D_ / 2 + 255) / 256, 256, 0, stream>>>(
        FV + (size_t)ix * D_ * D_, FVp + (size_t)ix * D_ * D_ / 2, D_, D_);
  }
  pack_b<<<(D_ * DOUT_ / 2 + 255) / 256, 256, 0, stream>>>(Wc, Wcp, D_, DOUT_);

  // ---- input projections (WMMA over bf16 features) ------------------------
  conv_launch(feat_user, fub, (size_t)NU * DIN_, stream);
  conv_launch(feat_item, fib, (size_t)NI * DIN_, stream);
  gemm_launch(fub, DIN_, WinUp, bin_user, hu, D_, NU, D_, DIN_, stream);
  gemm_launch(fib, DIN_, WinIp, bin_item, hi, D_, NI, D_, DIN_, stream);

  for (int l = 0; l < L_; ++l) {
    int i0 = l * 2 + 0, i1 = l * 2 + 1;
    // bf16 activations for this layer's 6 projection GEMMs
    conv_launch(hu, hub, (size_t)NU * D_, stream);
    conv_launch(hi, hib, (size_t)NI * D_, stream);

    gemm_launch(hub, D_, Wqp + (size_t)i0 * D_ * D_ / 2, bq + (size_t)i0 * D_,
                qu, D_, NU, D_, D_, stream);
    gemm_launch(hub, D_, FKp + (size_t)i0 * D_ * D_ / 2, fbk + (size_t)i0 * D_,
                kpu, D_, NU, D_, D_, stream);
    gemm_launch(hub, D_, FVp + (size_t)i0 * D_ * D_ / 2, fbv + (size_t)i0 * D_,
                vpu, D_, NU, D_, D_, stream);
    gemm_launch(hib, D_, Wqp + (size_t)i1 * D_ * D_ / 2, bq + (size_t)i1 * D_,
                qi, D_, NI, D_, D_, stream);
    gemm_launch(hib, D_, FKp + (size_t)i1 * D_ * D_ / 2, fbk + (size_t)i1 * D_,
                kpi, D_, NI, D_, D_, stream);
    gemm_launch(hib, D_, FVp + (size_t)i1 * D_ * D_ / 2, fbv + (size_t)i1 * D_,
                vpi, D_, NI, D_, D_, stream);

    // hu/hi are dead now; reuse them as the aggregation targets
    fill_f32<<<((size_t)NU * D_ + 255) / 256, 256, 0, stream>>>(hu, 0.0f, (size_t)NU * D_);
    fill_f32<<<((size_t)NI * D_ + 255) / 256, 256, 0, stream>>>(hi, 0.0f, (size_t)NI * D_);

    // relation 0: user -> item (dst = item), aggregate into hi
    fill_u32<<<((size_t)NI * H_ + 255) / 256, 256, 0, stream>>>(mmax, 0u, (size_t)NI * H_);
    fill_f32<<<((size_t)NI * H_ + 255) / 256, 256, 0, stream>>>(den, 0.0f, (size_t)NI * H_);
    edge_score<<<((size_t)E * H_ + 255) / 256, 256, 0, stream>>>(
        kpu, qi, src_ui, dst_ui, mu + (size_t)i0 * H_, sc, mmax, E);
    edge_expsum<<<((size_t)E * H_ + 255) / 256, 256, 0, stream>>>(sc, mmax, dst_ui, den, E);
    edge_agg<<<((size_t)E * 64 + 255) / 256, 256, 0, stream>>>(sc, den, vpu, src_ui, dst_ui, hi, E);

    // relation 1: item -> user (dst = user), aggregate into hu
    fill_u32<<<((size_t)NU * H_ + 255) / 256, 256, 0, stream>>>(mmax, 0u, (size_t)NU * H_);
    fill_f32<<<((size_t)NU * H_ + 255) / 256, 256, 0, stream>>>(den, 0.0f, (size_t)NU * H_);
    edge_score<<<((size_t)E * H_ + 255) / 256, 256, 0, stream>>>(
        kpi, qu, src_iu, dst_iu, mu + (size_t)i1 * H_, sc, mmax, E);
    edge_expsum<<<((size_t)E * H_ + 255) / 256, 256, 0, stream>>>(sc, mmax, dst_iu, den, E);
    edge_agg<<<((size_t)E * 64 + 255) / 256, 256, 0, stream>>>(sc, den, vpi, src_iu, dst_iu, hu, E);
  }

  // ---- classifier on user nodes (WMMA, NT=1 since DOUT=16) ----------------
  conv_launch(hu, hub, (size_t)NU * D_, stream);
  gemm_launch(hub, D_, Wcp, bc, (float*)d_out, DOUT_, NU, DOUT_, D_, stream);
}
